// GQAttention_46995532153573
// MI455X (gfx1250) — compile-verified, matrix operand order checked by reading, not run on hardware
//
#include <hip/hip_runtime.h>
#include <hip/hip_bf16.h>
#include <math.h>

// ---------------------------------------------------------------------------
// GQA attention (B=1, T=2048, D=2048, NH=32, NKV=8, DH=64) for gfx1250.
// All matmuls on v_wmma_f32_16x16x32_f16 (f16 in, f32 accumulate).
// GEMMs stage the block-shared B tile in LDS via the gfx1250 async
// Global->LDS path (ASYNCcnt) when the builtins are available.
// ---------------------------------------------------------------------------

#define T_ 2048
#define D_ 2048
#define NH_ 32
#define NKV_ 8
#define DH_ 64
#define QD_ (NH_ * DH_)   // 2048
#define KD_ (NKV_ * DH_)  // 512

typedef __attribute__((ext_vector_type(16))) _Float16 v16h;
typedef __attribute__((ext_vector_type(8)))  _Float16 v8h;
typedef __attribute__((ext_vector_type(8)))  float    v8f;
typedef __attribute__((ext_vector_type(4)))  unsigned int v4u;
typedef __attribute__((ext_vector_type(4)))  int v4i;

union V16U { v16h v; v8h h[2]; };

#if defined(__gfx1250__) && __has_builtin(__builtin_amdgcn_global_load_async_to_lds_b128) && __has_builtin(__builtin_amdgcn_s_wait_asynccnt)
#define USE_ASYNC_LDS 1
#else
#define USE_ASYNC_LDS 0
#endif

#define AS1 __attribute__((address_space(1)))
#define AS3 __attribute__((address_space(3)))

// WMMA 16x16x32 f16 -> f32 wrapper (8-arg form, probe-verified).
__device__ __forceinline__ v8f wmma_f32_16x16x32_f16(v16h a, v16h b, v8f c) {
  return __builtin_amdgcn_wmma_f32_16x16x32_f16(
      /*neg_a=*/false, a, /*neg_b=*/false, b,
      /*c_mod=*/(short)0, c, /*reuse_a=*/false, /*reuse_b=*/false);
}

// Load a 16-bit A/B fragment for WMMA 16x16x32.
// ISA layout (CDNA5 ISA): lane l holds row (l&15), K = {half*8..half*8+7,
// 16+half*8..16+half*8+7} where half = l>>4.  With a row-major (or
// transposed-B) source this is exactly two contiguous 16-byte loads per lane.
__device__ __forceinline__ v16h load_frag16(const _Float16* __restrict__ base,
                                            int row, int ld, int k0, int half) {
  const _Float16* p = base + (size_t)row * ld + k0 + half * 8;
  V16U u;
  u.h[0] = *(const v8h*)(p);
  u.h[1] = *(const v8h*)(p + 16);
  return u.v;
}

// ---------------------------------------------------------------------------
// Elementwise converters
// ---------------------------------------------------------------------------
__global__ void k_f32_to_f16(const float* __restrict__ in,
                             _Float16* __restrict__ out, int n) {
  int i = blockIdx.x * blockDim.x + threadIdx.x;
  if (i < n) out[i] = (_Float16)in[i];
}

// in: R x C (f32, row-major) -> out: C x R (f16)  (i.e. out = in^T)
__global__ void k_transpose_to_f16(const float* __restrict__ in,
                                   _Float16* __restrict__ out, int R, int C) {
  int c = blockIdx.x * blockDim.x + threadIdx.x;
  int r = blockIdx.y;
  if (c < C) out[(size_t)c * R + r] = (_Float16)in[(size_t)r * C + c];
}

// ---------------------------------------------------------------------------
// Tiled WMMA GEMM: C[M,N] = A[M,K] (f16, row-major) * B[K,N], with B given as
// BT[N,K] (f16, row-major).  Block = 256 threads = 8 waves; block tile is
// 128x64 (each wave owns a 16x64 strip, A-fragment reused across 4 N-tiles).
// The 64x32 B tile is shared by all 8 waves and staged in LDS, double
// buffered; the copy uses GLOBAL_LOAD_ASYNC_TO_LDS_B128 (ASYNCcnt) when
// available, else a plain load + ds_store.
// ---------------------------------------------------------------------------
__global__ void __launch_bounds__(256)
k_gemm_wmma(const _Float16* __restrict__ A,
            const _Float16* __restrict__ BT,
            _Float16* __restrict__ C16,
            float* __restrict__ C32,
            int M, int N, int K) {
  const int tid  = threadIdx.x;
  const int wave = tid >> 5;
  const int lane = tid & 31;
  const int half = lane >> 4;
  const int lr   = lane & 15;

  const int m0 = blockIdx.y * 128 + wave * 16;
  const int n0 = blockIdx.x * 64;

  // Double-buffered shared B tile: [buf][n (64)][k (32)] f16 = 2 x 4KB.
  __shared__ _Float16 Bs[2][64][32];

  // Copy duty: 64 rows x 32 halfs = 256 x 16B chunks -> one b128 per thread.
  const int crow = tid >> 2;        // 0..63  (row within BT tile)
  const int ccol = (tid & 3) * 8;   // 0,8,16,24 (half offset within row)

  const _Float16* Abase = A + (size_t)m0 * K;
  const int nk = K / 32;

  auto issue_copy = [&](int buf, int k0) {
    const _Float16* gsrc = BT + (size_t)(n0 + crow) * K + k0 + ccol;
#if USE_ASYNC_LDS
    __builtin_amdgcn_global_load_async_to_lds_b128(
        (AS1 v4i*)gsrc, (AS3 v4i*)&Bs[buf][crow][ccol], 0, 0);
#else
    v4u tmp = *(const v4u*)gsrc;
    *(v4u*)&Bs[buf][crow][ccol] = tmp;
#endif
  };

  issue_copy(0, 0);

  v8f acc[4] = {};
  for (int kt = 0; kt < nk; ++kt) {
    const int cur = kt & 1;
    const int k0  = kt * 32;
    const bool more = (kt + 1) < nk;
    if (more) issue_copy(cur ^ 1, k0 + 32);

#if USE_ASYNC_LDS
    if (more) __builtin_amdgcn_s_wait_asynccnt(1);
    else      __builtin_amdgcn_s_wait_asynccnt(0);
#endif
    __syncthreads();  // B tile visible to all waves

    if (k0 + 256 < K)  // prefetch upcoming A rows (global_prefetch_b8)
      __builtin_prefetch(Abase + (size_t)lr * K + k0 + 256, 0, 1);

    v16h a = load_frag16(Abase, lr, K, k0, half);
#pragma unroll
    for (int j = 0; j < 4; ++j) {
      v16h b = load_frag16(&Bs[cur][0][0], j * 16 + lr, 32, 0, half);
      acc[j] = wmma_f32_16x16x32_f16(a, b, acc[j]);
    }
    __syncthreads();  // all waves done reading before this buffer is refilled
  }

  // C-frag layout: reg i -> row m = half*8+i, lane -> col = lr (per 16-col tile)
#pragma unroll
  for (int j = 0; j < 4; ++j) {
    int col = n0 + j * 16 + lr;
#pragma unroll
    for (int i = 0; i < 8; ++i) {
      size_t idx = (size_t)(m0 + half * 8 + i) * N + col;
      if (C32) C32[idx] = acc[j][i];
      else     C16[idx] = (_Float16)acc[j][i];
    }
  }
}

// ---------------------------------------------------------------------------
// RoPE + repack to head-major: src[t][h*64+dh] -> dst[h][t][dh], with
// out[..,:32] = h1*c - h2*s ; out[..,32:] = h1*s + h2*c ; optional 1/sqrt(DH)
// folded in (Q only).  One wave per (t, h); lane = rotary pair index.
// ---------------------------------------------------------------------------
__global__ void k_rope_pack(const _Float16* __restrict__ src,
                            _Float16* __restrict__ dst,
                            const float* __restrict__ rc,
                            const float* __restrict__ rs,
                            int heads, float scale) {
  int t = blockIdx.x;
  int h = blockIdx.y;
  int d = threadIdx.x;  // 0..31
  float c = rc[t * 32 + d];
  float s = rs[t * 32 + d];
  const _Float16* p = src + (size_t)t * heads * DH_ + h * DH_;
  float h1 = (float)p[d];
  float h2 = (float)p[d + 32];
  _Float16* q = dst + ((size_t)h * T_ + t) * DH_;
  q[d]      = (_Float16)((h1 * c - h2 * s) * scale);
  q[d + 32] = (_Float16)((h1 * s + h2 * c) * scale);
}

// V repack+transpose: src[t][g*64+dh] -> Vt[g][dh][t]  (makes the PV B-frag
// two contiguous 16B loads per lane)
__global__ void k_pack_vt(const _Float16* __restrict__ src,
                          _Float16* __restrict__ vt) {
  int t = blockIdx.x;
  int g = blockIdx.y;
  int d = threadIdx.x;  // 0..63
  vt[((size_t)g * DH_ + d) * T_ + t] = src[(size_t)t * KD_ + g * DH_ + d];
}

// ---------------------------------------------------------------------------
// Flash attention (causal, GQA).  Grid = (T/64, NH); block = 128 (4 waves).
// Wave w owns query rows q0..q0+15.  Key loop in 32-wide tiles:
//   S(16x32) = Q(16x64) Kt^T  -> 2 WMMAs per 16-col half
//   online softmax (row stats via 16-lane xor-shuffles within each wave half)
//   P -> LDS (C-layout) -> A-fragment reload ; O(16x64) += P V  (4 WMMAs)
// ---------------------------------------------------------------------------
__global__ void k_flash_attn(const _Float16* __restrict__ Qh,  // [NH][T][DH]
                             const _Float16* __restrict__ Kh,  // [NKV][T][DH]
                             const _Float16* __restrict__ Vt,  // [NKV][DH][T]
                             _Float16* __restrict__ Y) {       // [T][NH*DH]
  const int h    = blockIdx.y;
  const int g    = h >> 2;  // REP = NH/NKV = 4
  const int wave = threadIdx.x >> 5;
  const int lane = threadIdx.x & 31;
  const int half = lane >> 4;
  const int lr   = lane & 15;
  const int q0   = blockIdx.x * 64 + wave * 16;

  const _Float16* Qbase = Qh + ((size_t)h * T_ + q0) * DH_;
  const _Float16* Kbase = Kh + (size_t)g * T_ * DH_;
  const _Float16* Vbase = Vt + (size_t)g * DH_ * T_;

  __shared__ _Float16 Pst[4][16][32];  // per-wave P staging (C->A relayout)

  // Q fragments for the full DH=64 (two k-steps), loaded once.
  v16h qa0 = load_frag16(Qbase, lr, DH_, 0, half);
  v16h qa1 = load_frag16(Qbase, lr, DH_, 32, half);

  v8f o[4] = {};
  float mrow[8], lrow[8];
#pragma unroll
  for (int i = 0; i < 8; ++i) { mrow[i] = -1e30f; lrow[i] = 0.0f; }

  const int send = q0 + 16;  // causal: keys 0 .. q0+15
  for (int s0 = 0; s0 < send; s0 += 32) {
    // ---- S = Q K^T over a 16x32 key tile (two 16x16 C-frags) ----
    v8f sf[2];
#pragma unroll
    for (int fi = 0; fi < 2; ++fi) {
      const _Float16* Kt = Kbase + (size_t)(s0 + fi * 16) * DH_;
      v8f z = {};
      v16h kb0 = load_frag16(Kt, lr, DH_, 0, half);
      v16h kb1 = load_frag16(Kt, lr, DH_, 32, half);
      z = wmma_f32_16x16x32_f16(qa0, kb0, z);
      z = wmma_f32_16x16x32_f16(qa1, kb1, z);
      sf[fi] = z;
    }

    // ---- causal mask + online softmax; write P (f16) into LDS ----
#pragma unroll
    for (int i = 0; i < 8; ++i) {
      const int row = q0 + half * 8 + i;
      float s0v = sf[0][i];
      float s1v = sf[1][i];
      if (s0 + lr > row)      s0v = -1e30f;
      if (s0 + 16 + lr > row) s1v = -1e30f;

      float mx = fmaxf(s0v, s1v);
#pragma unroll
      for (int d = 1; d < 16; d <<= 1) mx = fmaxf(mx, __shfl_xor(mx, d, 32));

      float mnew  = fmaxf(mrow[i], mx);
      float alpha = __expf(mrow[i] - mnew);
      float p0 = __expf(s0v - mnew);
      float p1 = __expf(s1v - mnew);

      float rsum = p0 + p1;
#pragma unroll
      for (int d = 1; d < 16; d <<= 1) rsum += __shfl_xor(rsum, d, 32);

      lrow[i] = lrow[i] * alpha + rsum;
      mrow[i] = mnew;
#pragma unroll
      for (int j = 0; j < 4; ++j) o[j][i] *= alpha;

      Pst[wave][half * 8 + i][lr]      = (_Float16)p0;
      Pst[wave][half * 8 + i][16 + lr] = (_Float16)p1;
    }

    // ---- reload P as an A-fragment (wave-local LDS round trip) ----
    const v8h* prow = (const v8h*)&Pst[wave][lr][0];  // 4 x v8h per row
    V16U pu;
    pu.h[0] = prow[half];      // K = half*8 .. half*8+7
    pu.h[1] = prow[2 + half];  // K = 16+half*8 ..
    v16h pa = pu.v;

    // ---- O += P V : B-frag b[i] = V[k][n] = Vt[n][s0+k] (contiguous) ----
#pragma unroll
    for (int j = 0; j < 4; ++j) {
      v16h vb = load_frag16(Vbase, j * 16 + lr, T_, s0, half);
      o[j] = wmma_f32_16x16x32_f16(pa, vb, o[j]);
    }
  }

  // ---- normalize and store Y[t][h*64+dh] ----
#pragma unroll
  for (int i = 0; i < 8; ++i) {
    float inv = 1.0f / lrow[i];
    size_t rowoff = (size_t)(q0 + half * 8 + i) * QD_ + h * DH_;
#pragma unroll
    for (int j = 0; j < 4; ++j)
      Y[rowoff + j * 16 + lr] = (_Float16)(o[j][i] * inv);
  }
}

// ---------------------------------------------------------------------------
// Host-side pipeline
// ---------------------------------------------------------------------------
extern "C" void kernel_launch(void* const* d_in, const int* in_sizes, int n_in,
                              void* d_out, int out_size, void* d_ws, size_t ws_size,
                              hipStream_t stream) {
  const float* x  = (const float*)d_in[0];
  const float* rc = (const float*)d_in[1];
  const float* rs = (const float*)d_in[2];
  const float* Wq = (const float*)d_in[3];
  const float* Wk = (const float*)d_in[4];
  const float* Wv = (const float*)d_in[5];
  const float* Wo = (const float*)d_in[6];
  float* out = (float*)d_out;

  char* ws = (char*)d_ws;
  size_t off = 0;
  auto alloc = [&](size_t bytes) -> _Float16* {
    _Float16* p = (_Float16*)(ws + off);
    off += (bytes + 255) & ~(size_t)255;
    return p;
  };
  _Float16* xh  = alloc((size_t)T_ * D_ * 2);
  _Float16* WqT = alloc((size_t)QD_ * D_ * 2);
  _Float16* WkT = alloc((size_t)KD_ * D_ * 2);
  _Float16* WvT = alloc((size_t)KD_ * D_ * 2);
  _Float16* WoT = alloc((size_t)D_ * QD_ * 2);
  _Float16* Q0  = alloc((size_t)T_ * QD_ * 2);
  _Float16* K0  = alloc((size_t)T_ * KD_ * 2);
  _Float16* V0  = alloc((size_t)T_ * KD_ * 2);
  _Float16* Qhm = alloc((size_t)NH_ * T_ * DH_ * 2);
  _Float16* Khm = alloc((size_t)NKV_ * T_ * DH_ * 2);
  _Float16* Vtm = alloc((size_t)NKV_ * DH_ * T_ * 2);
  _Float16* Y   = alloc((size_t)T_ * QD_ * 2);

  // 1) precision conversion + weight transposes (f32 -> f16)
  {
    int n = T_ * D_;
    k_f32_to_f16<<<(n + 255) / 256, 256, 0, stream>>>(x, xh, n);
  }
  k_transpose_to_f16<<<dim3((QD_ + 255) / 256, D_), 256, 0, stream>>>(Wq, WqT, D_, QD_);
  k_transpose_to_f16<<<dim3((KD_ + 255) / 256, D_), 256, 0, stream>>>(Wk, WkT, D_, KD_);
  k_transpose_to_f16<<<dim3((KD_ + 255) / 256, D_), 256, 0, stream>>>(Wv, WvT, D_, KD_);
  k_transpose_to_f16<<<dim3((D_ + 255) / 256, QD_), 256, 0, stream>>>(Wo, WoT, QD_, D_);

  // 2) projections (WMMA GEMMs; 128x64 block tiles, LDS-staged B)
  k_gemm_wmma<<<dim3(QD_ / 64, T_ / 128), 256, 0, stream>>>(xh, WqT, Q0, nullptr, T_, QD_, D_);
  k_gemm_wmma<<<dim3(KD_ / 64, T_ / 128), 256, 0, stream>>>(xh, WkT, K0, nullptr, T_, KD_, D_);
  k_gemm_wmma<<<dim3(KD_ / 64, T_ / 128), 256, 0, stream>>>(xh, WvT, V0, nullptr, T_, KD_, D_);

  // 3) RoPE (+ 1/sqrt(DH) folded into Q) and head-major repacks
  const float scale = 0.125f;  // 1/sqrt(64)
  k_rope_pack<<<dim3(T_, NH_), 32, 0, stream>>>(Q0, Qhm, rc, rs, NH_, scale);
  k_rope_pack<<<dim3(T_, NKV_), 32, 0, stream>>>(K0, Khm, rc, rs, NKV_, 1.0f);
  k_pack_vt<<<dim3(T_, NKV_), 64, 0, stream>>>(V0, Vtm);

  // 4) causal flash attention
  k_flash_attn<<<dim3(T_ / 64, NH_), 128, 0, stream>>>(Qhm, Khm, Vtm, Y);

  // 5) output projection (f32 result straight to d_out)
  k_gemm_wmma<<<dim3(D_ / 64, T_ / 128), 256, 0, stream>>>(Y, WoT, nullptr, out, T_, D_, QD_);
}
